// CausalSelfAttention_39213051412899
// MI455X (gfx1250) — matrix operand order repaired and checked by reading, not compiled, not run on hardware
//
#include <hip/hip_runtime.h>

typedef __attribute__((ext_vector_type(16))) __bf16 v16bf;
typedef __attribute__((ext_vector_type(8)))  float  v8f;

#define ZERO8 {0.f,0.f,0.f,0.f,0.f,0.f,0.f,0.f}

__device__ __forceinline__ v8f wmma_bf16(v16bf a, v16bf b, v8f c) {
  return __builtin_amdgcn_wmma_f32_16x16x32_bf16(false, a, false, b, (short)0, c, false, false);
}

// ---------------- prep: x fp32 -> bf16 A-frags (frag = 32 lanes x 16 bf16 = 1KB) ----
__global__ __launch_bounds__(256) void prep_x_kernel(const float* __restrict__ x,
                                                     __bf16* __restrict__ xf) {
  int wave = (blockIdx.x * 256 + threadIdx.x) >> 5;   // 0..8191 = rt*32 + kt
  int lane = threadIdx.x & 31;
  int rt = wave >> 5, kt = wave & 31;
  int row = rt * 16 + (lane & 15);
  int col = kt * 32 + (lane >> 4) * 8;
  const float* src = x + (size_t)row * 1024 + col;
  v16bf o;
#pragma unroll
  for (int i = 0; i < 8; ++i) { o[i] = (__bf16)src[i]; o[8 + i] = (__bf16)src[16 + i]; }
  *(v16bf*)(xf + (size_t)wave * 512 + lane * 16) = o;
}

// ---------------- prep: W fp32 [K][N] -> bf16 B-frags (lane = N col, elem = K) -----
__global__ __launch_bounds__(256) void prep_w_kernel(const float* __restrict__ w,
                                                     __bf16* __restrict__ wf, int NT) {
  int wave = (blockIdx.x * 256 + threadIdx.x) >> 5;   // kt*NT + nt
  int lane = threadIdx.x & 31;
  if (wave >= 32 * NT) return;
  int kt = wave / NT, nt = wave % NT;
  const float* src = w + (size_t)(kt * 32 + (lane >> 4) * 8) * (NT * 16) + nt * 16 + (lane & 15);
  v16bf o;
#pragma unroll
  for (int i = 0; i < 8; ++i) { o[i] = (__bf16)src[(size_t)i * (NT * 16)]; o[8 + i] = (__bf16)src[(size_t)(16 + i) * (NT * 16)]; }
  *(v16bf*)(wf + (size_t)wave * 512 + lane * 16) = o;
}

// ---------------- QKV + gate + RoPE + RMS; emits q B-frags, k A-frags, V^T A-frags --
__global__ __launch_bounds__(256) void qkv_kernel(
    const float* __restrict__ x, const float* __restrict__ ve,
    const float* __restrict__ cosp, const float* __restrict__ sinp,
    const float* __restrict__ wgate,
    const __bf16* __restrict__ xf, const __bf16* __restrict__ wqf,
    const __bf16* __restrict__ wkf, const __bf16* __restrict__ wvf,
    __bf16* __restrict__ qf, __bf16* __restrict__ kf, __bf16* __restrict__ vtf) {
  __shared__ __attribute__((aligned(32))) __bf16 vbufT[4 * 64 * 32]; // [kv][d][32 tok]
  __shared__ __attribute__((aligned(32))) __bf16 qkbuf[8][16 * 64];  // per-wave [tok][d]
  __shared__ float gate_lds[32 * 4];

  const int bb = blockIdx.x;            // 0..127 : covers 32 tokens
  const int b  = bb >> 6;
  const int s0 = (bb & 63) * 32;
  const int tid = threadIdx.x;
  const int lane = tid & 31;
  const int w = tid >> 5;
  const int lo = lane & 15;
  const int hi = lane >> 4;

  { // gate = 2*sigmoid(x[:, :128] @ Wgate) — 2 threads per (token, kv)
    int pair = tid >> 1, half = tid & 1;
    int tok = pair >> 2, kv = pair & 3;
    const float* xr = x + (size_t)(b * 2048 + s0 + tok) * 1024 + half * 64;
    float acc = 0.f;
#pragma unroll 4
    for (int i = 0; i < 64; ++i) acc += xr[i] * wgate[(half * 64 + i) * 4 + kv];
    acc += __shfl_xor(acc, 1, 32);
    if (half == 0) gate_lds[tok * 4 + kv] = 2.0f / (1.0f + __expf(-acc));
  }
  __syncthreads();

  // 48 GEMM units (8 v, 8 k, 32 q), 6 per wave. Each: 16 rows x 64 cols, K=1024.
  for (int ui = 0; ui < 6; ++ui) {
    int u = w * 6 + ui;
    int kind, rt, hh;
    if (u < 8)       { kind = 2; rt = u >> 2;        hh = u & 3; }        // v
    else if (u < 16) { kind = 1; rt = (u - 8) >> 2;  hh = (u - 8) & 3; }  // k
    else             { kind = 0; rt = (u - 16) >> 4; hh = (u - 16) & 15; } // q
    const int rt_g = bb * 2 + rt;
    const __bf16* wf = (kind == 0) ? wqf : ((kind == 1) ? wkf : wvf);
    const int NT = (kind == 0) ? 64 : 16;
    const int ntb = hh * 4;
    v8f a0 = ZERO8, a1 = ZERO8, a2 = ZERO8, a3 = ZERO8;
    for (int kt = 0; kt < 32; ++kt) {
      v16bf afr = *(const v16bf*)(xf + ((size_t)rt_g * 32 + kt) * 512 + lane * 16);
      const __bf16* wb = wf + ((size_t)kt * NT + ntb) * 512 + lane * 16;
      a0 = wmma_bf16(afr, *(const v16bf*)(wb),        a0);
      a1 = wmma_bf16(afr, *(const v16bf*)(wb + 512),  a1);
      a2 = wmma_bf16(afr, *(const v16bf*)(wb + 1024), a2);
      a3 = wmma_bf16(afr, *(const v16bf*)(wb + 1536), a3);
    }
    if (kind == 2) {                    // v = gemm + gate * ve  -> vbufT[kv][d][tok]
      const int kv = hh;
#pragma unroll
      for (int r = 0; r < 8; ++r) {
        int tokl = rt * 16 + r + hi * 8;
        float g = gate_lds[tokl * 4 + kv];
        const float* vp = ve + (size_t)(b * 2048 + s0 + tokl) * 256 + kv * 64 + lo;
        __bf16* vb = vbufT + kv * 2048 + lo * 32 + tokl;
        vb[0 * 512] = (__bf16)(a0[r] + g * vp[0]);
        vb[1 * 512] = (__bf16)(a1[r] + g * vp[16]);
        vb[2 * 512] = (__bf16)(a2[r] + g * vp[32]);
        vb[3 * 512] = (__bf16)(a3[r] + g * vp[48]);
      }
    } else {                            // q/k: rope + rms, stage in per-wave LDS
#pragma unroll
      for (int r = 0; r < 8; ++r) {
        int tokl16 = r + hi * 8;
        int pos = s0 + rt * 16 + tokl16;
        const float* cp = cosp + (size_t)pos * 32;
        const float* sp = sinp + (size_t)pos * 32;
        float c0 = cp[lo], c1 = cp[16 + lo];
        float n0 = sp[lo], n1 = sp[16 + lo];
        float q0 = a0[r], q1 = a1[r], q2 = a2[r], q3 = a3[r];
        float o0 =  q0 * c0 + q2 * n0;
        float o1 =  q1 * c1 + q3 * n1;
        float o2 = -q0 * n0 + q2 * c0;
        float o3 = -q1 * n1 + q3 * c1;
        float ss = o0 * o0 + o1 * o1 + o2 * o2 + o3 * o3;
        ss += __shfl_xor(ss, 1, 32); ss += __shfl_xor(ss, 2, 32);
        ss += __shfl_xor(ss, 4, 32); ss += __shfl_xor(ss, 8, 32);
        float sc = rsqrtf(ss * (1.0f / 64.0f) + 1.1920929e-07f);
        __bf16* qb = qkbuf[w] + tokl16 * 64;
        qb[lo]      = (__bf16)(o0 * sc);
        qb[16 + lo] = (__bf16)(o1 * sc);
        qb[32 + lo] = (__bf16)(o2 * sc);
        qb[48 + lo] = (__bf16)(o3 * sc);
      }
      if (kind == 0) {                  // q -> B-frag of Q^T: lane = tok col, elem = d
        size_t base = (((size_t)(b * 16 + hh) * 128 + ((s0 >> 4) + rt)) * 2) * 512;
#pragma unroll
        for (int ks = 0; ks < 2; ++ks) {
          const __bf16* src = qkbuf[w] + lo * 64 + ks * 32 + hi * 8;
          v16bf o;
#pragma unroll
          for (int e = 0; e < 8; ++e) { o[e] = src[e]; o[8 + e] = src[16 + e]; }
          *(v16bf*)(qf + base + ks * 512 + lane * 16) = o;
        }
      } else {                          // k -> A-frag: lane = 16+tok split, elem = d
        size_t base = (((size_t)(b * 4 + hh) * 128 + ((s0 >> 4) + rt)) * 2) * 512;
#pragma unroll
        for (int ks = 0; ks < 2; ++ks) {
          const __bf16* src = qkbuf[w] + lo * 64 + ks * 32 + hi * 8;
          v16bf o;
#pragma unroll
          for (int i = 0; i < 8; ++i) { o[i] = src[i]; o[8 + i] = src[16 + i]; }
          *(v16bf*)(kf + base + ks * 512 + lane * 16) = o;
        }
      }
    }
  }
  __syncthreads();
  // V^T A-frags (M = d tile of 16, K = 32 keys): 16 frags, 2 per wave
#pragma unroll
  for (int i = 0; i < 2; ++i) {
    int f = w * 2 + i;
    int kv = f >> 2, dt = f & 3;
    const __bf16* src = vbufT + kv * 2048 + (dt * 16 + lo) * 32 + hi * 8;
    v16bf o;
#pragma unroll
    for (int j = 0; j < 8; ++j) { o[j] = src[j]; o[8 + j] = src[16 + j]; }
    size_t base = (((size_t)(b * 4 + kv) * 64 + (bb & 63)) * 4 + dt) * 512;
    *(v16bf*)(vtf + base + lane * 16) = o;
  }
}

// ---------------- flash attention, transposed scores S^T = K * Q^T -----------------
__global__ __launch_bounds__(256) void attn_kernel(
    const __bf16* __restrict__ qf, const __bf16* __restrict__ kf,
    const __bf16* __restrict__ vtf, __bf16* __restrict__ yf,
    const int* __restrict__ wlp, const int* __restrict__ wrp) {
  __shared__ __attribute__((aligned(32))) __bf16 pbuf[8][32 * 16]; // per-wave P^T, stored [q][32 j]
  const int gw = blockIdx.x * 8 + (threadIdx.x >> 5);  // (b,h,qt)
  const int lane = threadIdx.x & 31;
  const int w = threadIdx.x >> 5;
  const int lo = lane & 15, hi = lane >> 4;
  const int qt = gw & 127;
  const int h = (gw >> 7) & 15;
  const int b = gw >> 11;
  const int kv = h >> 2;
  const int wl = wlp[0], wr = wrp[0];
  const int qi0 = qt * 16;
  const int qq = qi0 + lo;              // this lane's query index

  size_t qbase = (((size_t)(b * 16 + h) * 128 + qt) * 2) * 512;
  v16bf qfr0 = *(const v16bf*)(qf + qbase + lane * 16);
  v16bf qfr1 = *(const v16bf*)(qf + qbase + 512 + lane * 16);

  v8f acc0 = ZERO8, acc1 = ZERO8, acc2 = ZERO8, acc3 = ZERO8;
  float m_run = -1e30f, l_run = 0.f;

  int start = qi0 - wl; if (start < 0) start = 0;
  const int jt_lo = start >> 5;
  const int jt_hi = (qi0 + 15) >> 5;
  __bf16* pb = pbuf[w];
  for (int jt = jt_lo; jt <= jt_hi; ++jt) {
    const int jb = jt * 32;
    const __bf16* kb = kf + (((size_t)(b * 4 + kv) * 128 + jt * 2) * 2) * 512 + lane * 16;
    v8f st0 = ZERO8, st1 = ZERO8;
    st0 = wmma_bf16(*(const v16bf*)(kb),        qfr0, st0);
    st0 = wmma_bf16(*(const v16bf*)(kb + 512),  qfr1, st0);
    st1 = wmma_bf16(*(const v16bf*)(kb + 1024), qfr0, st1);
    st1 = wmma_bf16(*(const v16bf*)(kb + 1536), qfr1, st1);
    float p[16];
    float mx = -1e30f;
#pragma unroll
    for (int r = 0; r < 8; ++r) {
      int j0 = jb + hi * 8 + r;
      int j1 = j0 + 16;
      float v0 = st0[r] * 0.125f;       // D^-0.5
      float v1 = st1[r] * 0.125f;
      bool k0 = (j0 <= qq + wr) && (j0 + wl >= qq) && (j0 <= qq);
      bool k1 = (j1 <= qq + wr) && (j1 + wl >= qq) && (j1 <= qq);
      p[r]     = k0 ? v0 : -1e30f;
      p[8 + r] = k1 ? v1 : -1e30f;
      mx = fmaxf(mx, fmaxf(p[r], p[8 + r]));
    }
    mx = fmaxf(mx, __shfl_xor(mx, 16, 32));
    float m_new = fmaxf(m_run, mx);
    float alpha = __expf(m_run - m_new);
    float rsum = 0.f;
#pragma unroll
    for (int i = 0; i < 16; ++i) {
      float pe = (p[i] < -1e29f) ? 0.f : __expf(p[i] - m_new);
      p[i] = pe; rsum += pe;
    }
    rsum += __shfl_xor(rsum, 16, 32);
    l_run = l_run * alpha + rsum;
    m_run = m_new;
#pragma unroll
    for (int r = 0; r < 8; ++r) { acc0[r] *= alpha; acc1[r] *= alpha; acc2[r] *= alpha; acc3[r] *= alpha; }
    // P^T through per-wave LDS: D-layout -> B-frag (lane = q col, elem = j)
#pragma unroll
    for (int r = 0; r < 8; ++r) {
      pb[(lo * 2 + hi) * 16 + r]      = (__bf16)p[r];
      pb[(lo * 2 + hi) * 16 + 8 + r] = (__bf16)p[8 + r];
    }
    v16bf pfr = *(const v16bf*)(pb + (lo * 2 + hi) * 16);
    const __bf16* vb = vtf + (((size_t)(b * 4 + kv) * 64 + jt) * 4) * 512 + lane * 16;
    acc0 = wmma_bf16(*(const v16bf*)(vb),        pfr, acc0);
    acc1 = wmma_bf16(*(const v16bf*)(vb + 512),  pfr, acc1);
    acc2 = wmma_bf16(*(const v16bf*)(vb + 1024), pfr, acc2);
    acc3 = wmma_bf16(*(const v16bf*)(vb + 1536), pfr, acc3);
  }
  float invl = 1.0f / l_run;
  v16bf y0, y1;                         // lane-local repack: Y^T D-layout -> proj A-frags
#pragma unroll
  for (int r = 0; r < 8; ++r) {
    y0[r]     = (__bf16)(acc0[r] * invl);
    y0[8 + r] = (__bf16)(acc1[r] * invl);
    y1[r]     = (__bf16)(acc2[r] * invl);
    y1[8 + r] = (__bf16)(acc3[r] * invl);
  }
  size_t rt = (size_t)b * 128 + qt;
  *(v16bf*)(yf + (rt * 32 + h * 2) * 512 + lane * 16)     = y0;
  *(v16bf*)(yf + (rt * 32 + h * 2 + 1) * 512 + lane * 16) = y1;
}

// ---------------- out = y @ Wproj (fp32 out) ---------------------------------------
__global__ __launch_bounds__(256) void proj_kernel(
    const __bf16* __restrict__ yf, const __bf16* __restrict__ wpf,
    float* __restrict__ out) {
  const int gw = blockIdx.x * 8 + (threadIdx.x >> 5);
  const int lane = threadIdx.x & 31;
  const int lo = lane & 15, hi = lane >> 4;
  const int rt = gw >> 4;               // 0..255 row tiles
  const int ntg = gw & 15;              // group of 4 n-tiles
  v8f a0 = ZERO8, a1 = ZERO8, a2 = ZERO8, a3 = ZERO8;
  for (int kt = 0; kt < 32; ++kt) {
    v16bf afr = *(const v16bf*)(yf + ((size_t)rt * 32 + kt) * 512 + lane * 16);
    const __bf16* wb = wpf + ((size_t)kt * 64 + ntg * 4) * 512 + lane * 16;
    a0 = wmma_bf16(afr, *(const v16bf*)(wb),        a0);
    a1 = wmma_bf16(afr, *(const v16bf*)(wb + 512),  a1);
    a2 = wmma_bf16(afr, *(const v16bf*)(wb + 1024), a2);
    a3 = wmma_bf16(afr, *(const v16bf*)(wb + 1536), a3);
  }
#pragma unroll
  for (int r = 0; r < 8; ++r) {
    float* o = out + ((size_t)rt * 16 + r + hi * 8) * 1024 + ntg * 64 + lo;
    o[0] = a0[r]; o[16] = a1[r]; o[32] = a2[r]; o[48] = a3[r];
  }
}

extern "C" void kernel_launch(void* const* d_in, const int* in_sizes, int n_in,
                              void* d_out, int out_size, void* d_ws, size_t ws_size,
                              hipStream_t stream) {
  (void)in_sizes; (void)n_in; (void)out_size; (void)ws_size;
  const float* x    = (const float*)d_in[0];
  const float* ve   = (const float*)d_in[1];
  const float* cosp = (const float*)d_in[2];
  const float* sinp = (const float*)d_in[3];
  const float* Wq   = (const float*)d_in[4];
  const float* Wk   = (const float*)d_in[5];
  const float* Wv   = (const float*)d_in[6];
  const float* Wp   = (const float*)d_in[7];
  const float* Wg   = (const float*)d_in[8];
  const int*   wlp  = (const int*)d_in[9];
  const int*   wrp  = (const int*)d_in[10];
  float* out = (float*)d_out;

  char* ws = (char*)d_ws;
  size_t off = 0;
  __bf16* xf  = (__bf16*)(ws + off); off += (size_t)8192 * 1024; // x A-frags
  __bf16* wqf = (__bf16*)(ws + off); off += (size_t)2048 * 1024; // Wq B-frags
  __bf16* wkf = (__bf16*)(ws + off); off += (size_t)512  * 1024; // Wk B-frags
  __bf16* wvf = (__bf16*)(ws + off); off += (size_t)512  * 1024; // Wv B-frags
  __bf16* wpf = (__bf16*)(ws + off); off += (size_t)2048 * 1024; // Wproj B-frags
  __bf16* qfb = (__bf16*)(ws + off); off += (size_t)8192 * 1024; // Q^T B-frags
  __bf16* kfb = (__bf16*)(ws + off); off += (size_t)2048 * 1024; // K A-frags
  __bf16* vtf = (__bf16*)(ws + off); off += (size_t)2048 * 1024; // V^T A-frags
  __bf16* yfb = (__bf16*)(ws + off); off += (size_t)8192 * 1024; // y A-frags

  prep_x_kernel<<<1024, 256, 0, stream>>>(x, xf);
  prep_w_kernel<<<256, 256, 0, stream>>>(Wq, wqf, 64);
  prep_w_kernel<<<64,  256, 0, stream>>>(Wk, wkf, 16);
  prep_w_kernel<<<64,  256, 0, stream>>>(Wv, wvf, 16);
  prep_w_kernel<<<256, 256, 0, stream>>>(Wp, wpf, 64);
  qkv_kernel<<<128, 256, 0, stream>>>(x, ve, cosp, sinp, Wg, xf, wqf, wkf, wvf,
                                      qfb, kfb, vtf);
  attn_kernel<<<512, 256, 0, stream>>>(qfb, kfb, vtf, yfb, wlp, wrp);
  proj_kernel<<<512, 256, 0, stream>>>(yfb, wpf, out);
}
